// LSTM_model_34342558498965
// MI455X (gfx1250) — matrix operand order repaired and checked by reading, not compiled, not run on hardware
//
#include <hip/hip_runtime.h>
#include <hip/hip_bf16.h>
#include <math.h>

// ---------------------------------------------------------------------------
// Types for CDNA5 WMMA (wave32)
// ---------------------------------------------------------------------------
typedef __bf16 bf16_t;
typedef __attribute__((ext_vector_type(8)))  __bf16 v8bf;
typedef __attribute__((ext_vector_type(16))) __bf16 v16bf;
typedef __attribute__((ext_vector_type(8)))  float  v8f;

#define LSTM_B   32
#define LSTM_D   512
#define LSTM_T   1024
#define LSTM_H   512
#define NWG      16          // persistent workgroups; each owns 32 h-columns

// float -> bf16 round-to-nearest-even
static __device__ inline bf16_t f2bf(float f) {
    unsigned u = __float_as_uint(f);
    u += 0x7FFFu + ((u >> 16) & 1u);
    unsigned short s = (unsigned short)(u >> 16);
    bf16_t r;
    __builtin_memcpy(&r, &s, 2);
    return r;
}

static __device__ inline float fast_sigmoid(float x) {
    return 1.0f / (1.0f + __expf(-x));
}
static __device__ inline float fast_tanh(float x) {
    return 1.0f - 2.0f / (__expf(2.0f * x) + 1.0f);
}

// A-matrix (16x32 bf16) fragment load from row-major [.., ld] buffer.
// lane<16: row = m0+lane, K = [k0, k0+8) and [k0+16, k0+24)
// lane>=16: row = m0+lane-16, K = [k0+8, k0+16) and [k0+24, k0+32)
static __device__ inline v16bf load_a(const bf16_t* __restrict__ base, int ld,
                                      int m0, int k0, int lane) {
    const int row = m0 + (lane & 15);
    const int s   = (lane & 16) ? 8 : 0;
    const bf16_t* p = base + (size_t)row * ld + k0 + s;
    v8bf lo = *(const v8bf*)(p);
    v8bf hi = *(const v8bf*)(p + 16);
    v16bf a;
#pragma unroll
    for (int i = 0; i < 8; ++i) { a[i] = lo[i]; a[8 + i] = hi[i]; }
    return a;
}

// B-matrix (32x16 bf16) fragment load. B = W^T, so column n of B is row n of W
// (row-major, ld = 512). lane holds K = [k0+8h, k0+8h+8) and [k0+16+8h, k0+24+8h)
// for column lane&15, h = lane>>4.
static __device__ inline v16bf load_b(const bf16_t* __restrict__ wrow0, int k0, int lane) {
    const int n = lane & 15;
    const int s = (lane & 16) ? 8 : 0;
    const bf16_t* p = wrow0 + (size_t)n * LSTM_H + k0 + s;
    v8bf lo = *(const v8bf*)(p);
    v8bf hi = *(const v8bf*)(p + 16);
    v16bf b;
#pragma unroll
    for (int i = 0; i < 8; ++i) { b[i] = lo[i]; b[8 + i] = hi[i]; }
    return b;
}

static __device__ inline v8f wmma_bf16(v16bf a, v16bf b, v8f c) {
    return __builtin_amdgcn_wmma_f32_16x16x32_bf16(false, a, false, b,
                                                   (short)0, c, false, false);
}

// x-projection for one timestep: acc = bias + x_t @ Wih^T (this wave's 2 tiles)
static __device__ inline void xproj_acc(const bf16_t* __restrict__ xt,
                                        const bf16_t* __restrict__ wihRow,
                                        int mt, int lane,
                                        float bias0, float bias1,
                                        v8f& a0, v8f& a1) {
#pragma unroll
    for (int r = 0; r < 8; ++r) { a0[r] = bias0; a1[r] = bias1; }
#pragma unroll 4
    for (int k0 = 0; k0 < LSTM_D; k0 += 32) {
        v16bf a  = load_a(xt, LSTM_D, mt * 16, k0, lane);
        v16bf b0 = load_b(wihRow, k0, lane);
        v16bf b1 = load_b(wihRow + (size_t)16 * LSTM_H, k0, lane);
        a0 = wmma_bf16(a, b0, a0);
        a1 = wmma_bf16(a, b1, a1);
    }
}

// ---------------------------------------------------------------------------
// Kernel 1: weight/bias conversion + state init (re-run every launch so the
// barrier counter and h0 are deterministic across graph replays)
// ---------------------------------------------------------------------------
__global__ __launch_bounds__(256) void lstm_prep(
    const float* __restrict__ Wih, const float* __restrict__ Whh,
    const float* __restrict__ bih, const float* __restrict__ bhh,
    bf16_t* __restrict__ WihB, bf16_t* __restrict__ WhhB,
    float* __restrict__ bias, bf16_t* __restrict__ hbuf,
    unsigned* __restrict__ bar) {
    const int i = blockIdx.x * blockDim.x + threadIdx.x;
    if (i < 4 * LSTM_H * LSTM_D) {          // 2048*512
        WihB[i] = f2bf(Wih[i]);
        WhhB[i] = f2bf(Whh[i]);
    }
    if (i < 4 * LSTM_H) bias[i] = bih[i] + bhh[i];
    if (i < LSTM_B * LSTM_H) hbuf[i] = f2bf(0.0f);
    if (i == 0) *bar = 0u;
}

// ---------------------------------------------------------------------------
// Kernel 2: x [B,D,T] fp32 -> xT [T,B,D] bf16 (tiled LDS transpose)
// grid (T/32, D/32, B), block (32, 8)
// ---------------------------------------------------------------------------
__global__ __launch_bounds__(256) void lstm_xpose(const float* __restrict__ x,
                                                  bf16_t* __restrict__ xT) {
    __shared__ float tile[32][33];
    const int b  = blockIdx.z;
    const int d0 = blockIdx.y * 32;
    const int t0 = blockIdx.x * 32;
    const int tx = threadIdx.x, ty = threadIdx.y;
    const float* xb = x + (size_t)b * LSTM_D * LSTM_T;
#pragma unroll
    for (int j = 0; j < 4; ++j) {
        const int d = d0 + ty + 8 * j;
        tile[ty + 8 * j][tx] = xb[(size_t)d * LSTM_T + t0 + tx];
    }
    __syncthreads();
#pragma unroll
    for (int j = 0; j < 4; ++j) {
        const int t = t0 + ty + 8 * j;
        xT[(size_t)t * (LSTM_B * LSTM_D) + (size_t)b * LSTM_D + d0 + tx] =
            f2bf(tile[tx][ty + 8 * j]);
    }
}

// ---------------------------------------------------------------------------
// Kernel 3: persistent recurrent scan with split-barrier overlap.
// 16 WGs x 256 threads (8 waves). WG wg owns h columns [wg*32, wg*32+32).
// Wave w: gate = w&3, M-tile = w>>2; computes two 16x16 C tiles (N width 32).
// Critical path per step: barrier -> h@Whh^T (WMMA) -> cell -> arrive.
// The x_t@Wih^T projection for step t+1 is computed BETWEEN arrive and wait,
// hiding it (and its L2 latency) under the device-wide sync window.
// ---------------------------------------------------------------------------
__global__ __launch_bounds__(256) void lstm_persist(
    const bf16_t* __restrict__ xT, const bf16_t* __restrict__ Wih,
    const bf16_t* __restrict__ Whh, const float* __restrict__ bias,
    bf16_t* __restrict__ hbuf, unsigned* __restrict__ bar,
    float* __restrict__ out) {
    __shared__ float gbuf[4][32][32];   // [gate][b][n_local]
    __shared__ float cbuf[32][32];      // resident cell state slice

    const int tid  = threadIdx.x;
    const int lane = tid & 31;
    const int wave = tid >> 5;
    const int wg   = blockIdx.x;
    const int gate = wave & 3;
    const int mt   = wave >> 2;         // 0 or 1 (rows 0-15 / 16-31 of B=32)
    const int n0h  = wg * 32;           // this WG's h-column base

    for (int i = tid; i < 32 * 32; i += 256) (&cbuf[0][0])[i] = 0.0f;
    __syncthreads();

    const int nglob0 = gate * LSTM_H + n0h;              // gate-column base, nt=0
    const bf16_t* wihRow = Wih + (size_t)nglob0 * LSTM_H;
    const bf16_t* whhRow = Whh + (size_t)nglob0 * LSTM_H;
    const float bias0 = bias[nglob0 + (lane & 15)];
    const float bias1 = bias[nglob0 + 16 + (lane & 15)];

    // Prologue: x-projection for t=0 (h0 == 0, but we still run the full
    // h-GEMM in-loop for uniformity; hbuf is zeroed by lstm_prep).
    v8f accX0, accX1;
    xproj_acc(xT, wihRow, mt, lane, bias0, bias1, accX0, accX1);

    for (int t = 0; t < LSTM_T; ++t) {
        v8f acc0 = accX0, acc1 = accX1;

        // Prefetch next timestep's A tile while doing the recurrent GEMM.
        if (t + 1 < LSTM_T) {
            const bf16_t* xn = xT + (size_t)(t + 1) * (LSTM_B * LSTM_D)
                             + (size_t)(mt * 16 + (lane & 15)) * LSTM_D;
            __builtin_prefetch(xn, 0, 3);
        }

        // Recurrent projection: A = h (32x512), B = Whh^T slice
#pragma unroll 4
        for (int k0 = 0; k0 < LSTM_H; k0 += 32) {
            v16bf a  = load_a(hbuf, LSTM_H, mt * 16, k0, lane);
            v16bf b0 = load_b(whhRow, k0, lane);
            v16bf b1 = load_b(whhRow + (size_t)16 * LSTM_H, k0, lane);
            acc0 = wmma_bf16(a, b0, acc0);
            acc1 = wmma_bf16(a, b1, acc1);
        }

        // Scatter C tiles to LDS: element (m,n) at vgpr=m%8, lane=(m/8)*16+n
        {
            const int nl = lane & 15;
            const int mr = (lane >> 4) * 8;
#pragma unroll
            for (int r = 0; r < 8; ++r) {
                gbuf[gate][mt * 16 + mr + r][nl]      = acc0[r];
                gbuf[gate][mt * 16 + mr + r][16 + nl] = acc1[r];
            }
        }
        __syncthreads();

        // Fused LSTM cell: 32x32 elements over 256 threads
#pragma unroll
        for (int i = 0; i < 4; ++i) {
            const int idx = tid + i * 256;
            const int m = idx >> 5;
            const int n = idx & 31;
            const float gi = fast_sigmoid(gbuf[0][m][n]);
            const float gf = fast_sigmoid(gbuf[1][m][n]);
            const float gg = fast_tanh(gbuf[2][m][n]);
            const float go = fast_sigmoid(gbuf[3][m][n]);
            const float c  = gf * cbuf[m][n] + gi * gg;
            const float h  = go * fast_tanh(c);
            cbuf[m][n] = c;
            out[(size_t)t * (LSTM_B * LSTM_H) + (size_t)m * LSTM_H + n0h + n] = h;
            hbuf[(size_t)m * LSTM_H + n0h + n] = f2bf(h);
        }

        // ---- split device-wide barrier ----
        __threadfence();            // release: h stores visible at agent scope
        __syncthreads();
        if (tid == 0) {             // arrive
            __hip_atomic_fetch_add(bar, 1u, __ATOMIC_RELEASE,
                                   __HIP_MEMORY_SCOPE_AGENT);
        }

        // Overlap: x-projection for t+1 (independent of other WGs' h stores)
        if (t + 1 < LSTM_T) {
            const bf16_t* xn = xT + (size_t)(t + 1) * (LSTM_B * LSTM_D);
            xproj_acc(xn, wihRow, mt, lane, bias0, bias1, accX0, accX1);
        }

        if (tid == 0) {             // wait
            const unsigned target = (unsigned)(t + 1) * NWG;
            while (__hip_atomic_load(bar, __ATOMIC_ACQUIRE,
                                     __HIP_MEMORY_SCOPE_AGENT) < target) {
                __builtin_amdgcn_s_sleep(2);
            }
        }
        __syncthreads();
        __threadfence();            // acquire: don't read stale h from WGP$
    }
}

// ---------------------------------------------------------------------------
// Host launcher. Workspace layout (bytes):
//   [0, 32MB)        xT  bf16  [T,B,D]
//   [32MB, 34MB)     Wih bf16  [2048,512]
//   [34MB, 36MB)     Whh bf16  [2048,512]
//   [36MB, +8KB)     bias fp32 [2048]
//   [36MB+64KB, ..)  hbuf bf16 [32,512]
//   [36MB+128KB]     barrier counter
// ---------------------------------------------------------------------------
extern "C" void kernel_launch(void* const* d_in, const int* in_sizes, int n_in,
                              void* d_out, int out_size, void* d_ws, size_t ws_size,
                              hipStream_t stream) {
    const float* x   = (const float*)d_in[0];
    const float* Wih = (const float*)d_in[1];
    const float* Whh = (const float*)d_in[2];
    const float* bih = (const float*)d_in[3];
    const float* bhh = (const float*)d_in[4];
    float* out = (float*)d_out;

    char* ws = (char*)d_ws;
    bf16_t*   xT   = (bf16_t*)(ws);
    bf16_t*   WihB = (bf16_t*)(ws + ((size_t)32 << 20));
    bf16_t*   WhhB = (bf16_t*)(ws + ((size_t)34 << 20));
    float*    bias = (float*) (ws + ((size_t)36 << 20));
    bf16_t*   hbuf = (bf16_t*)(ws + ((size_t)36 << 20) + (64u << 10));
    unsigned* bar  = (unsigned*)(ws + ((size_t)36 << 20) + (128u << 10));

    // Phase 1: weight conversion + state init (covers 2048*512 elements)
    lstm_prep<<<(4 * LSTM_H * LSTM_D + 255) / 256, 256, 0, stream>>>(
        Wih, Whh, bih, bhh, WihB, WhhB, bias, hbuf, bar);

    // Phase 2: x transpose/convert to time-major bf16
    dim3 tb(32, 8);
    dim3 tg(LSTM_T / 32, LSTM_D / 32, LSTM_B);
    lstm_xpose<<<tg, tb, 0, stream>>>(x, xT);

    // Phase 3: persistent recurrent scan
    lstm_persist<<<NWG, 256, 0, stream>>>(xT, WihB, WhhB, bias, hbuf, bar, out);
}
